// GraphAttentionLayer_63299228009044
// MI455X (gfx1250) — matrix operand order repaired and checked by reading, not compiled, not run on hardware
//
#include <hip/hip_runtime.h>
#include <hip/hip_bf16.h>

typedef __attribute__((ext_vector_type(16))) _Float16 v16h;
typedef __attribute__((ext_vector_type(8)))  _Float16 v8h;
typedef __attribute__((ext_vector_type(8)))  float    v8f;

#define NN    4096
#define INF_  256
#define OUTF  64
#define NH    8

// ---------------- workspace layout (bytes) ----------------
#define OFF_XH   ((size_t)0)                     // x f16:   4096*256*2      = 2 MB
#define OFF_WT   (OFF_XH  + (size_t)2097152)     // W^T f16: 8*64*256*2      = 256 KB
#define OFF_H32  (OFF_WT  + (size_t)262144)      // h f32:   8*4096*64*4     = 8 MB
#define OFF_HT   (OFF_H32 + (size_t)8388608)     // h^T f16: 8*64*4096*2     = 4 MB
#define OFF_SRC  (OFF_HT  + (size_t)4194304)     // src f32: 8*4096*4        = 128 KB
#define OFF_DST  (OFF_SRC + (size_t)131072)      // dst f32
#define OFF_MX   (OFF_DST + (size_t)131072)      // row max f32
#define OFF_BITS (OFF_MX  + (size_t)131072)      // adj bitmask: 4096*128*4  = 2 MB
#define OFF_HP   (OFF_BITS+ (size_t)2097152)     // per-head h': 8*4096*64*4 = 8 MB

// ---------------- conversions ----------------
__global__ void gat_conv_x(const float* __restrict__ x, _Float16* __restrict__ xh) {
  int i = blockIdx.x * blockDim.x + threadIdx.x;           // 4096*256
  xh[i] = (_Float16)x[i];
}

// wt[(h*64+o)*256 + k] = W[(h*256+k)*64 + o]   (transpose so B-tile K is contiguous)
__global__ void gat_conv_wt(const float* __restrict__ W, _Float16* __restrict__ wt) {
  int i = blockIdx.x * blockDim.x + threadIdx.x;           // 8*64*256
  int k = i & 255, o = (i >> 8) & 63, h = i >> 14;
  wt[i] = (_Float16)W[((h << 8) + k) * 64 + o];
}

// ---------------- h = x @ W  (per head), WMMA f16->f32 ----------------
__global__ __launch_bounds__(32)
void gat_gemm_h(const _Float16* __restrict__ xh, const _Float16* __restrict__ wt,
                float* __restrict__ h32) {
  const int lane = threadIdx.x;
  const int n0   = blockIdx.x * 16;
  const int hh   = blockIdx.y;
  const int m    = lane & 15;                 // A-matrix row within tile
  const int akof = (lane < 16) ? 0 : 16;      // A K offset per half-wave
  const int bkof = (lane < 16) ? 0 : 16;      // B K offset per half-wave
  v8f acc[4];
  const v8f zero = {0.f,0.f,0.f,0.f,0.f,0.f,0.f,0.f};
#pragma unroll
  for (int t = 0; t < 4; ++t) acc[t] = zero;

  for (int kc = 0; kc < INF_; kc += 32) {
    const _Float16* ap = xh + (size_t)(n0 + m) * INF_ + kc + akof;
    v8h lo = *(const v8h*)ap;
    v8h hi = *(const v8h*)(ap + 8);
    v16h A = __builtin_shufflevector(lo, hi, 0,1,2,3,4,5,6,7,8,9,10,11,12,13,14,15);
#pragma unroll
    for (int t = 0; t < 4; ++t) {
      int o = t * 16 + m;
      v16h B = *(const v16h*)(wt + ((size_t)(hh * 64 + o) * INF_) + kc + bkof);
      acc[t] = __builtin_amdgcn_wmma_f32_16x16x32_f16(false, A, false, B,
                                                      (short)0, acc[t], false, false);
    }
  }
  // C layout: VGPR i -> row (i | lane>=16 ? +8), lane&15 -> col
#pragma unroll
  for (int i = 0; i < 8; ++i) {
    int n = n0 + i + ((lane < 16) ? 0 : 8);
#pragma unroll
    for (int t = 0; t < 4; ++t) {
      int o = t * 16 + m;
      h32[((size_t)hh * NN + n) * OUTF + o] = acc[t][i];
    }
  }
}

// hT[(h*64+o)*4096 + n] = f16(h32[(h*4096+n)*64 + o])
__global__ void gat_transpose_h(const float* __restrict__ h32, _Float16* __restrict__ hT) {
  int i = blockIdx.x * blockDim.x + threadIdx.x;           // 8*64*4096, n fastest
  int n = i & 4095, o = (i >> 12) & 63, h = i >> 18;
  hT[i] = (_Float16)h32[((size_t)(h << 12) + n) * OUTF + o];
}

// src[h][n] = h[h,n,:].a[h,:64] ; dst uses a[h,64:128]
__global__ void gat_srcdst(const float* __restrict__ h32, const float* __restrict__ a,
                           float* __restrict__ src, float* __restrict__ dst) {
  int i = blockIdx.x * blockDim.x + threadIdx.x;           // 8*4096
  int n = i & 4095, h = i >> 12;
  const float* hp = h32 + ((size_t)(h << 12) + n) * OUTF;
  const float* av = a + h * 128;
  float s = 0.f, d = 0.f;
#pragma unroll
  for (int o = 0; o < OUTF; ++o) { s += hp[o] * av[o]; d += hp[o] * av[64 + o]; }
  src[i] = s; dst[i] = d;
}

// pack adj (int32 0/1) into bitmasks, one wave32 ballot per 32 entries
__global__ void gat_adjpack(const int* __restrict__ adj, unsigned* __restrict__ bits) {
  size_t i = (size_t)blockIdx.x * blockDim.x + threadIdx.x; // 4096*4096
  unsigned long long b = __ballot(adj[i] != 0);
  if ((threadIdx.x & 31) == 0) bits[i >> 5] = (unsigned)b;
}

// mx[h][n] = leaky(src[h][n] + max_{adj(n,m)} dst[h][m])
__global__ __launch_bounds__(32)
void gat_rowmax(const unsigned* __restrict__ bits, const float* __restrict__ dst,
                const float* __restrict__ src, float* __restrict__ mx) {
  const int n = blockIdx.x, lane = threadIdx.x;
  float m8[NH];
#pragma unroll
  for (int h = 0; h < NH; ++h) m8[h] = -3.4e38f;
  for (int w = lane; w < 128; w += 32) {
    unsigned msk = bits[(size_t)n * 128 + w];
    while (msk) {
      int b = __ffs((int)msk) - 1; msk &= msk - 1;
      int m = w * 32 + b;
#pragma unroll
      for (int h = 0; h < NH; ++h) m8[h] = fmaxf(m8[h], dst[(h << 12) + m]);
    }
  }
#pragma unroll
  for (int off = 16; off >= 1; off >>= 1)
#pragma unroll
    for (int h = 0; h < NH; ++h) m8[h] = fmaxf(m8[h], __shfl_xor(m8[h], off, 32));
  if (lane == 0) {
#pragma unroll
    for (int h = 0; h < NH; ++h) {
      float t = src[(h << 12) + n] + m8[h];
      float e = (t > 0.f) ? t : 0.2f * t;
      if (m8[h] < -1e37f) e = 0.f;              // isolated row guard
      mx[(h << 12) + n] = e;
    }
  }
}

// Fused masked-softmax + alpha@h : one wave per (head, 16-row tile)
__global__ __launch_bounds__(32)
void gat_attn(const unsigned* __restrict__ bits, const float* __restrict__ src,
              const float* __restrict__ dst, const float* __restrict__ mx,
              const _Float16* __restrict__ hT, float* __restrict__ hp) {
  __shared__ float sdst[NN];
  __shared__ float sden[16];
  const int lane = threadIdx.x;
  const int n0   = blockIdx.x * 16;
  const int hh   = blockIdx.y;
  const int m    = lane & 15;
  const int nrow = n0 + m;
  const int kb   = (lane < 16) ? 0 : 16;      // A-layout K base per half-wave
  const int bkof = (lane < 16) ? 0 : 16;      // B-layout K offset per half-wave

  for (int i = lane; i < NN; i += 32) sdst[i] = dst[(hh << 12) + i];
  __syncthreads();

  const float s_n  = src[(hh << 12) + nrow];
  const float mx_n = mx [(hh << 12) + nrow];

  v8f acc[4];
  const v8f zero = {0.f,0.f,0.f,0.f,0.f,0.f,0.f,0.f};
#pragma unroll
  for (int t = 0; t < 4; ++t) acc[t] = zero;
  float denom = 0.f;

  const unsigned* brow = bits + (size_t)nrow * 128;
  for (int c = 0; c < 128; ++c) {
    const int m0 = c * 32;
    const unsigned msk = brow[c];
    v16h A;
#pragma unroll
    for (int j = 0; j < 16; ++j) {            // K = kb + j
      int k = kb + j;
      float t = s_n + sdst[m0 + k];
      float e = (t > 0.f) ? t : 0.2f * t;
      float p = ((msk >> k) & 1u) ? __expf(e - mx_n) : 0.f;
      denom += p;
      A[j] = (_Float16)p;
    }
    const _Float16* bp = hT + (size_t)(hh * 64) * NN + m0 + bkof;
#pragma unroll
    for (int t = 0; t < 4; ++t) {
      int o = t * 16 + m;
      v16h B = *(const v16h*)(bp + (size_t)o * NN);
      acc[t] = __builtin_amdgcn_wmma_f32_16x16x32_f16(false, A, false, B,
                                                      (short)0, acc[t], false, false);
    }
  }

  float tot = denom + __shfl_xor(denom, 16, 32);
  if (lane < 16) sden[lane] = tot;
  __syncthreads();

#pragma unroll
  for (int i = 0; i < 8; ++i) {
    int M = i + ((lane < 16) ? 0 : 8);
    float r = 1.0f / sden[M];
    int n = n0 + M;
#pragma unroll
    for (int t = 0; t < 4; ++t) {
      int o = t * 16 + m;
      hp[((size_t)hh * NN + n) * OUTF + o] = acc[t][i] * r;
    }
  }
}

// out = elu(mean over heads)
__global__ void gat_final(const float* __restrict__ hp, float* __restrict__ out) {
  int i = blockIdx.x * blockDim.x + threadIdx.x;           // 4096*64
  float s = 0.f;
#pragma unroll
  for (int h = 0; h < NH; ++h) s += hp[(size_t)h * NN * OUTF + i];
  float v = s * 0.125f;
  out[i] = (v > 0.f) ? v : (__expf(v) - 1.f);
}

extern "C" void kernel_launch(void* const* d_in, const int* in_sizes, int n_in,
                              void* d_out, int out_size, void* d_ws, size_t ws_size,
                              hipStream_t stream) {
  const float* x   = (const float*)d_in[0];
  const int*   adj = (const int*)  d_in[1];
  const float* W   = (const float*)d_in[2];
  const float* a   = (const float*)d_in[3];
  float* out = (float*)d_out;
  char* ws = (char*)d_ws;

  _Float16* xh   = (_Float16*)(ws + OFF_XH);
  _Float16* wt   = (_Float16*)(ws + OFF_WT);
  float*    h32  = (float*)   (ws + OFF_H32);
  _Float16* hT   = (_Float16*)(ws + OFF_HT);
  float*    src  = (float*)   (ws + OFF_SRC);
  float*    dst  = (float*)   (ws + OFF_DST);
  float*    mx   = (float*)   (ws + OFF_MX);
  unsigned* bits = (unsigned*)(ws + OFF_BITS);
  float*    hp   = (float*)   (ws + OFF_HP);

  gat_conv_x   <<<(NN*INF_)/256, 256, 0, stream>>>(x, xh);
  gat_conv_wt  <<<(NH*OUTF*INF_)/256, 256, 0, stream>>>(W, wt);
  gat_adjpack  <<<((size_t)NN*NN)/256, 256, 0, stream>>>(adj, bits);
  gat_gemm_h   <<<dim3(NN/16, NH), 32, 0, stream>>>(xh, wt, h32);
  gat_transpose_h<<<(NH*OUTF*NN)/256, 256, 0, stream>>>(h32, hT);
  gat_srcdst   <<<(NH*NN)/256, 256, 0, stream>>>(h32, a, src, dst);
  gat_rowmax   <<<NN, 32, 0, stream>>>(bits, dst, src, mx);
  gat_attn     <<<dim3(NN/16, NH), 32, 0, stream>>>(bits, src, dst, mx, hT, hp);
  gat_final    <<<(NN*OUTF)/256, 256, 0, stream>>>(hp, out);
}
